// Pattention_3977139716106
// MI455X (gfx1250) — matrix operand order repaired and checked by reading, not compiled, not run on hardware
//
#include <hip/hip_runtime.h>
#include <math.h>

// Problem constants (from reference)
static constexpr int kB  = 4;
static constexpr int kL  = 2048;
static constexpr int kD  = 1024;
static constexpr int kS  = 4096;
static constexpr int kDV = 1024;
static constexpr int kRows = kB * kL;          // 8192
static constexpr int kTM   = 32;               // rows per workgroup (2 WMMA M-tiles)
static constexpr int kWaves = 8;               // 256 threads / wave32

typedef _Float16 f16;
typedef __attribute__((ext_vector_type(16))) _Float16 v16h;
typedef __attribute__((ext_vector_type(8)))  _Float16 v8h;
typedef __attribute__((ext_vector_type(4)))  _Float16 v4h;
typedef __attribute__((ext_vector_type(8)))  float    v8f;

// Branchless exact-gelu: erf via Abramowitz–Stegun 7.1.26 (|err| < 1.5e-7),
// using the HW transcendental pipe (v_rcp_f32, v_exp_f32). No exec-mask churn.
__device__ __forceinline__ float gelu_exact(float x) {
    const float z  = 0.70710678118654752f * x;       // x / sqrt(2)
    const float az = fabsf(z);
    const float t  = __builtin_amdgcn_rcpf(fmaf(0.3275911f, az, 1.0f));
    float p = fmaf(1.061405429f, t, -1.453152027f);
    p = fmaf(p, t, 1.421413741f);
    p = fmaf(p, t, -0.284496736f);
    p = fmaf(p, t, 0.254829592f);
    p *= t;
    const float e = __builtin_amdgcn_exp2f(-az * az * 1.4426950408889634f);
    const float erf_abs = fmaf(-p, e, 1.0f);
    const float erf_z   = copysignf(erf_abs, z);
    return 0.5f * x * (1.0f + erf_z);
}

// XOR-swizzled LDS addressing on 8-half groups: lanes reading rows 0..15 at the
// same logical column hit 16 distinct bank quads (all 64 banks, conflict-free).
__device__ __forceinline__ int swz_x(int row, int col) {   // lds_x: row stride kD
    int g = col >> 3;
    g = (g & ~15) | ((g ^ row) & 15);
    return row * kD + (g << 3) + (col & 7);
}
__device__ __forceinline__ int swz_g(int row, int col) {   // lds_g: row stride kS
    int g = col >> 3;
    g = (g & ~15) | ((g ^ row) & 15);
    return row * kS + (g << 3) + (col & 7);
}

// ---------------- Pre-pass 1: K (f32) -> K16 (f16) ----------------
__global__ __launch_bounds__(256)
void cvt_f32_to_f16(const float* __restrict__ src, f16* __restrict__ dst, int n4) {
    const int i = blockIdx.x * 256 + threadIdx.x;
    if (i < n4) {
        const float4 f = reinterpret_cast<const float4*>(src)[i];
        v4h h;
        h[0] = (f16)f.x; h[1] = (f16)f.y; h[2] = (f16)f.z; h[3] = (f16)f.w;
        reinterpret_cast<v4h*>(dst)[i] = h;
    }
}

// ---------------- Pre-pass 2: V [S][DV] f32 -> VT [DV][S] f16 (LDS-tiled) ----------------
__global__ __launch_bounds__(256)
void transpose_cvt_f16(const float* __restrict__ V, f16* __restrict__ VT) {
    __shared__ f16 tile[64][65];
    const int n0 = blockIdx.x * 64;              // along DV
    const int s0 = blockIdx.y * 64;              // along S
    for (int idx = threadIdx.x; idx < 64 * 64; idx += 256) {
        const int r = idx >> 6, c = idx & 63;
        tile[r][c] = (f16)V[(size_t)(s0 + r) * kDV + n0 + c];
    }
    __syncthreads();
    for (int idx = threadIdx.x; idx < 64 * 64; idx += 256) {
        const int r = idx >> 6, c = idx & 63;
        VT[(size_t)(n0 + r) * kS + s0 + c] = tile[c][r];
    }
}

// ---------------- Fused Pattention ----------------
template <bool PRECONV>
__global__ __launch_bounds__(256)
void pattention_fused(const float* __restrict__ X,
                      const float* __restrict__ K,
                      const float* __restrict__ V,
                      const f16* __restrict__ K16,
                      const f16* __restrict__ VT16,
                      float* __restrict__ O) {
    extern __shared__ unsigned char smem_raw[];
    f16* lds_x = reinterpret_cast<f16*>(smem_raw);            // kTM x kD  (64 KB, swizzled)
    f16* lds_g = lds_x + kTM * kD;                            // kTM x kS  (256 KB, swizzled)
    float* lds_red = reinterpret_cast<float*>(lds_x);         // aliases lds_x (dead after ph.1)

    const int tid  = threadIdx.x;
    const int wave = tid >> 5;
    const int lane = tid & 31;
    const int l16  = lane & 15;
    const int hi   = lane >> 4;
    const int rowBase = blockIdx.x * kTM;

    // ---------- Phase 0: stage X tile (f32 -> f16, swizzled) into LDS ----------
    for (int idx = tid; idx < kTM * kD / 4; idx += 256) {
        const int r  = idx >> 8;                 // kD/4 = 256 float4 per row
        const int c4 = (idx & 255) << 2;
        const float4 f = *reinterpret_cast<const float4*>(X + (size_t)(rowBase + r) * kD + c4);
        v4h h;
        h[0] = (f16)f.x; h[1] = (f16)f.y; h[2] = (f16)f.z; h[3] = (f16)f.w;
        *reinterpret_cast<v4h*>(lds_x + swz_x(r, c4)) = h;   // c4%4==0: stays in one 8-group
    }
    __syncthreads();

    // ---------- Phase 1: scores = X_tile @ K^T (2 M-tiles share each B), gelu, sum(g^2) ----------
    float accsq0[8], accsq1[8];
    #pragma unroll
    for (int r = 0; r < 8; ++r) { accsq0[r] = 0.0f; accsq1[r] = 0.0f; }

    for (int cIdx = wave; cIdx < kS / 16; cIdx += kWaves) {
        const int sBase = cIdx * 16;
        if (cIdx + kWaves < kS / 16) {
            if (PRECONV) __builtin_prefetch(K16 + (size_t)(sBase + kWaves * 16 + l16) * kD, 0, 1);
            else         __builtin_prefetch(K   + (size_t)(sBase + kWaves * 16 + l16) * kD, 0, 1);
        }
        const float* krow   = K + (size_t)(sBase + l16) * kD;
        const f16*   krow16 = PRECONV ? (K16 + (size_t)(sBase + l16) * kD) : nullptr;

        v8f acc0 = {}, acc1 = {};
        #pragma unroll 2
        for (int k0 = 0; k0 < kD; k0 += 32) {
            // B fragment: column n = l16; lanes 0-15 K=0..15, lanes 16-31 K=16..31
            v16h b;
            if (PRECONV) {
                b = *reinterpret_cast<const v16h*>(krow16 + k0 + hi * 16);
            } else {
                const float4* kp = reinterpret_cast<const float4*>(krow + k0 + hi * 16);
                #pragma unroll
                for (int q = 0; q < 4; ++q) {
                    const float4 f = kp[q];
                    b[4 * q + 0] = (f16)f.x; b[4 * q + 1] = (f16)f.y;
                    b[4 * q + 2] = (f16)f.z; b[4 * q + 3] = (f16)f.w;
                }
            }
            // A fragments for both M-tiles (rows l16 and 16+l16), swizzled groups
            const int gb = (k0 >> 3) + 2 * hi;   // first 8-group of this lane's A slice
            v16h a0, a1;
            {
                const f16* p00 = lds_x + swz_x(l16,      gb << 3);
                const f16* p01 = lds_x + swz_x(l16,      (gb + 1) << 3);
                const f16* p10 = lds_x + swz_x(16 + l16, gb << 3);
                const f16* p11 = lds_x + swz_x(16 + l16, (gb + 1) << 3);
                #pragma unroll
                for (int i = 0; i < 8; ++i) { a0[i] = p00[i]; a0[8 + i] = p01[i]; }
                #pragma unroll
                for (int i = 0; i < 8; ++i) { a1[i] = p10[i]; a1[8 + i] = p11[i]; }
            }
            acc0 = __builtin_amdgcn_wmma_f32_16x16x32_f16(false, a0, false, b,
                                                          (short)0, acc0, false, false);
            acc1 = __builtin_amdgcn_wmma_f32_16x16x32_f16(false, a1, false, b,
                                                          (short)0, acc1, false, false);
        }
        // gelu + sumsq + stash g (swizzled) for both tiles; row = r+8*hi (+16)
        #pragma unroll
        for (int r = 0; r < 8; ++r) {
            const float g0 = gelu_exact(acc0[r]);
            const float g1 = gelu_exact(acc1[r]);
            accsq0[r] += g0 * g0;
            accsq1[r] += g1 * g1;
            lds_g[swz_g(r + 8 * hi,      sBase + l16)] = (f16)g0;
            lds_g[swz_g(16 + r + 8 * hi, sBase + l16)] = (f16)g1;
        }
    }
    __syncthreads();                              // lds_x dead -> lds_red may alias it
    if (tid < kTM) lds_red[tid] = 0.0f;
    __syncthreads();
    #pragma unroll
    for (int r = 0; r < 8; ++r) {
        atomicAdd(&lds_red[r + 8 * hi],      accsq0[r]);   // ds_add_f32
        atomicAdd(&lds_red[16 + r + 8 * hi], accsq1[r]);
    }
    __syncthreads();

    // ---------- Phase 2: scale = sqrt(S)*rsqrt(sumsq); rescale g in place (physical order) ----------
    if (tid < kTM) lds_red[tid] = 64.0f * rsqrtf(lds_red[tid]);   // sqrt(4096) = 64
    __syncthreads();
    for (int idx = tid; idx < kTM * kS / 8; idx += 256) {         // 8-half vectors, row-pure
        const int row = idx >> 9;                                  // (idx*8) / kS
        const float sc = lds_red[row];
        v8h* p = reinterpret_cast<v8h*>(lds_g) + idx;
        v8h h = *p;
        #pragma unroll
        for (int i = 0; i < 8; ++i) h[i] = (f16)((float)h[i] * sc);
        *p = h;
    }
    __syncthreads();

    // ---------- Phase 3: out tile = w @ V (2 M-tiles share each B) ----------
    for (int nIdx = wave; nIdx < kDV / 16; nIdx += kWaves) {
        const int nBase = nIdx * 16;
        const f16* vtrow = PRECONV ? (VT16 + (size_t)(nBase + l16) * kS) : nullptr;
        v8f acc0 = {}, acc1 = {};
        #pragma unroll 2
        for (int s0 = 0; s0 < kS; s0 += 32) {
            v16h b;
            if (PRECONV) {
                b = *reinterpret_cast<const v16h*>(vtrow + s0 + hi * 16);
            } else {
                const float* vp = V + (size_t)(s0 + hi * 16) * kDV + nBase + l16;
                #pragma unroll
                for (int i = 0; i < 16; ++i) b[i] = (f16)vp[(size_t)i * kDV];
            }
            const int gb = (s0 >> 3) + 2 * hi;
            v16h a0, a1;
            {
                const f16* p00 = lds_g + swz_g(l16,      gb << 3);
                const f16* p01 = lds_g + swz_g(l16,      (gb + 1) << 3);
                const f16* p10 = lds_g + swz_g(16 + l16, gb << 3);
                const f16* p11 = lds_g + swz_g(16 + l16, (gb + 1) << 3);
                #pragma unroll
                for (int i = 0; i < 8; ++i) { a0[i] = p00[i]; a0[8 + i] = p01[i]; }
                #pragma unroll
                for (int i = 0; i < 8; ++i) { a1[i] = p10[i]; a1[8 + i] = p11[i]; }
            }
            acc0 = __builtin_amdgcn_wmma_f32_16x16x32_f16(false, a0, false, b,
                                                          (short)0, acc0, false, false);
            acc1 = __builtin_amdgcn_wmma_f32_16x16x32_f16(false, a1, false, b,
                                                          (short)0, acc1, false, false);
        }
        #pragma unroll
        for (int r = 0; r < 8; ++r) {
            O[(size_t)(rowBase + r + 8 * hi) * kDV + nBase + l16]      = acc0[r];
            O[(size_t)(rowBase + 16 + r + 8 * hi) * kDV + nBase + l16] = acc1[r];
        }
    }
}

extern "C" void kernel_launch(void* const* d_in, const int* in_sizes, int n_in,
                              void* d_out, int out_size, void* d_ws, size_t ws_size,
                              hipStream_t stream) {
    (void)in_sizes; (void)n_in; (void)out_size;
    const float* X = (const float*)d_in[0];   // [B, L, D]
    const float* K = (const float*)d_in[1];   // [S, D]
    const float* V = (const float*)d_in[2];   // [S, DV]
    float* O = (float*)d_out;                 // [B, L, DV]

    const size_t shmem = (size_t)kTM * (kD + kS) * sizeof(_Float16);  // 320 KB (full WGP LDS)
    const size_t wsNeed = (size_t)kS * kD * sizeof(_Float16)
                        + (size_t)kS * kDV * sizeof(_Float16);        // 16 MB (L2-resident)

    if (ws_size >= wsNeed) {
        f16* K16  = (f16*)d_ws;
        f16* VT16 = K16 + (size_t)kS * kD;
        const int n4 = (kS * kD) / 4;
        cvt_f32_to_f16<<<(n4 + 255) / 256, 256, 0, stream>>>(K, K16, n4);
        transpose_cvt_f16<<<dim3(kDV / 64, kS / 64), 256, 0, stream>>>(V, VT16);
        hipFuncSetAttribute(reinterpret_cast<const void*>(&pattention_fused<true>),
                            hipFuncAttributeMaxDynamicSharedMemorySize, (int)shmem);
        pattention_fused<true><<<kRows / kTM, 256, shmem, stream>>>(X, K, V, K16, VT16, O);
    } else {
        hipFuncSetAttribute(reinterpret_cast<const void*>(&pattention_fused<false>),
                            hipFuncAttributeMaxDynamicSharedMemorySize, (int)shmem);
        pattention_fused<false><<<kRows / kTM, 256, shmem, stream>>>(X, K, V, nullptr, nullptr, O);
    }
}
